// QuantumMultiHeadAttention_65481071401651
// MI455X (gfx1250) — matrix operand order repaired and checked by reading, not compiled, not run on hardware
//
#include <hip/hip_runtime.h>
#include <hip/hip_bf16.h>

// ---------------------------------------------------------------------------
// Quantum multi-head attention for MI455X (gfx1250, wave32, WMMA).
//   B=4, N=2048, D=1024, H=16, HD=64
// Pipeline: qkv_pack (cos projections -> f16, V transposed)  ->
//           qattn_flash (fixed-shift softmax attention, wmma f16->f32) ->
//           out_proj (WMMA GEMM vs W_out^T + bias, f32 out)
//
// Softmax uses a FIXED shift of 0: scores = (q.k)/sqrt(64) with q,k = cos(..)
// are provably in [-8, 8], so exp(s) is in [e^-8, e^8=2981] -- safely inside
// f16/f32 range. Softmax is shift-invariant, so this is exact and removes the
// entire online-softmax machinery (no row max, no rescale, no per-iteration
// cross-lane reductions).
// ---------------------------------------------------------------------------

#define B_  4
#define N_  2048
#define D_  1024
#define H_  16
#define HD_ 64
#define BH_ (B_ * H_)

typedef __attribute__((ext_vector_type(16))) _Float16 v16h;
typedef __attribute__((ext_vector_type(8)))  _Float16 v8h;
typedef __attribute__((ext_vector_type(8)))  float    v8f;

__device__ __forceinline__ v8f vzero8() {
  v8f r;
#pragma unroll
  for (int i = 0; i < 8; ++i) r[i] = 0.0f;
  return r;
}

// Concatenate two contiguous 8-half (16B) loads into one 16-half fragment.
// Element i of the result lands in VGPR i/2 (low/high half), matching the
// ISA A/B fragment register packing.
__device__ __forceinline__ v16h ldfrag(const _Float16* p0, const _Float16* p1) {
  v8h lo = *(const v8h*)p0;
  v8h hi = *(const v8h*)p1;
  v16h r;
#pragma unroll
  for (int i = 0; i < 8; ++i) { r[i] = lo[i]; r[i + 8] = hi[i]; }
  return r;
}

__device__ __forceinline__ v8f wmma16x16x32(v16h a, v16h b, v8f c) {
  // D = A(16x32 f16) * B(32x16 f16) + C(16x16 f32)
  return __builtin_amdgcn_wmma_f32_16x16x32_f16(
      /*neg_a=*/false, a, /*neg_b=*/false, b,
      /*c_mod=*/(short)0, c, /*reuse_a=*/false, /*reuse_b=*/false);
}

// ---------------------------------------------------------------------------
// Kernel 1: q = cos(x+tq)/sqrt(HD), k = cos(x+tk), vt = cos(x+tv) transposed.
//   q,k : [BH, N, HD] f16      vt : [BH, HD, N] f16
// ---------------------------------------------------------------------------
__global__ __launch_bounds__(256) void qkv_pack(
    const float* __restrict__ x,
    const float* __restrict__ tq, const float* __restrict__ tk,
    const float* __restrict__ tv,
    _Float16* __restrict__ q, _Float16* __restrict__ k,
    _Float16* __restrict__ vt)
{
  size_t i = (size_t)blockIdx.x * 256 + threadIdx.x;   // over B*N*D
  int d  = (int)(i % D_);
  size_t bn = i / D_;
  int n  = (int)(bn % N_);
  int b  = (int)(bn / N_);
  int hh = d / HD_;
  int hd = d % HD_;
  float xv = x[i];
  int   th = hh * HD_ + hd;
  size_t rowmaj = (((size_t)b * H_ + hh) * N_ + n) * HD_ + hd;
  // fold softmax scale 1/sqrt(64)=0.125 into Q  =>  |score| <= 8
  q[rowmaj] = (_Float16)(__cosf(xv + tq[th]) * 0.125f);
  k[rowmaj] = (_Float16)(__cosf(xv + tk[th]));
  vt[(((size_t)b * H_ + hh) * HD_ + hd) * N_ + n] = (_Float16)(__cosf(xv + tv[th]));
}

__global__ __launch_bounds__(256) void w_pack(const float* __restrict__ w,
                                              _Float16* __restrict__ wh)
{
  size_t i = (size_t)blockIdx.x * 256 + threadIdx.x;   // over D*D
  wh[i] = (_Float16)w[i];
}

// ---------------------------------------------------------------------------
// Kernel 2: attention with fixed-shift softmax. One wave per 16 query rows;
// 8 waves / block. Per 32-key iteration:
//   4 WMMA (S = Q K^T)  ->  16 v_exp  ->  P via per-wave LDS (D->A layout)
//   ->  4 WMMA (O += P V).  Row sums accumulate per-lane; one cross-lane
// reduction at the end.
// ---------------------------------------------------------------------------
__global__ __launch_bounds__(256) void qattn_flash(
    const _Float16* __restrict__ q, const _Float16* __restrict__ k,
    const _Float16* __restrict__ vt, _Float16* __restrict__ o)
{
  __shared__ _Float16 pbuf[8 * 16 * 32];        // 8 KB: per-wave P tile

  const int lane  = threadIdx.x & 31;
  const int wave  = threadIdx.x >> 5;
  const int m     = lane & 15;                  // row (A frag) / col (B, C/D)
  const int hf    = lane >> 4;                  // lane half

  const int nTiles = N_ / 128;
  const int bh     = blockIdx.x / nTiles;
  const int qrow0  = (blockIdx.x % nTiles) * 128 + wave * 16;

  const _Float16* qb = q  + (size_t)bh * N_  * HD_;
  const _Float16* kb = k  + (size_t)bh * N_  * HD_;
  const _Float16* vb = vt + (size_t)bh * HD_ * N_;

  // Q A-fragments (held in registers for the whole key pass):
  //   aq0: hd 0..31, aq1: hd 32..63
  const _Float16* qp = qb + (size_t)(qrow0 + m) * HD_ + hf * 8;
  v16h aq0 = ldfrag(qp,      qp + 16);
  v16h aq1 = ldfrag(qp + 32, qp + 48);

  v8f oacc[4];
#pragma unroll
  for (int c = 0; c < 4; ++c) oacc[c] = vzero8();
  float lsum[8];
#pragma unroll
  for (int r = 0; r < 8; ++r) lsum[r] = 0.0f;

  _Float16* pw = pbuf + wave * (16 * 32);

  for (int k0 = 0; k0 < N_; k0 += 32) {
    // prefetch next key block (K rows + V rows) into the cache hierarchy
    {
      int kn = (k0 + 32) & (N_ - 1);
      __builtin_prefetch(kb + (size_t)(kn + m) * HD_ + hf * 16, 0, 1);
      __builtin_prefetch(vb + (size_t)(m) * N_ + kn + hf * 16, 0, 1);
    }

    // ---- S = Q * K^T (pre-scaled), two 16x16 column tiles ----
    v8f s[2];
#pragma unroll
    for (int t = 0; t < 2; ++t) {
      s[t] = vzero8();
      const _Float16* kp = kb + (size_t)(k0 + t * 16 + m) * HD_ + hf * 8;
      v16h bk0 = ldfrag(kp,      kp + 16);   // hd 0..31  slice
      v16h bk1 = ldfrag(kp + 32, kp + 48);   // hd 32..63 slice
      s[t] = wmma16x16x32(aq0, bk0, s[t]);
      s[t] = wmma16x16x32(aq1, bk1, s[t]);
    }

    // ---- fixed-shift softmax numerator: p = exp(s), s in [-8, 8] ----
#pragma unroll
    for (int r = 0; r < 8; ++r) {
      float p0 = __expf(s[0][r]);
      float p1 = __expf(s[1][r]);
      s[0][r] = p0; s[1][r] = p1;
      lsum[r] += p0 + p1;                    // per-lane partial row sums
    }

    // ---- P: C/D layout -> A layout via per-wave LDS staging ----
#pragma unroll
    for (int t = 0; t < 2; ++t)
#pragma unroll
      for (int r = 0; r < 8; ++r)
        pw[(r + 8 * hf) * 32 + t * 16 + m] = (_Float16)s[t][r];
    asm volatile("s_wait_dscnt 0" ::: "memory");
    const _Float16* pp = pw + m * 32 + hf * 8;
    v16h pa = ldfrag(pp, pp + 16);

    // ---- O += P * V  (V pre-transposed: contiguous B-fragments) ----
#pragma unroll
    for (int c = 0; c < 4; ++c) {
      const _Float16* vp = vb + (size_t)(c * 16 + m) * N_ + k0 + hf * 8;
      v16h bv = ldfrag(vp, vp + 16);
      oacc[c] = wmma16x16x32(pa, bv, oacc[c]);
    }
  }

  // ---- one final cross-lane row-sum reduction (cols live across the 16
  //      lanes of each half; rows = vgpr index + 8*hf) ----
  float inv[8];
#pragma unroll
  for (int r = 0; r < 8; ++r) {
    float sv = lsum[r];
    sv += __shfl_xor(sv, 1, 32);
    sv += __shfl_xor(sv, 2, 32);
    sv += __shfl_xor(sv, 4, 32);
    sv += __shfl_xor(sv, 8, 32);
    inv[r] = 1.0f / sv;
  }

  // ---- normalize and store O as f16 in [B, N, D] (= [B,N,H,HD]) ----
  const int bb = bh / H_;
  const int hh = bh % H_;
  _Float16* ob = o + (size_t)bb * N_ * D_ + hh * HD_;
#pragma unroll
  for (int c = 0; c < 4; ++c)
#pragma unroll
    for (int r = 0; r < 8; ++r) {
      int row = qrow0 + 8 * hf + r;
      int col = c * 16 + m;
      ob[(size_t)row * D_ + col] = (_Float16)(oacc[c][r] * inv[r]);
    }
}

// ---------------------------------------------------------------------------
// Kernel 3: out = attn_out[BN, D] @ W_out^T + b_out   (f32 output)
// One wave -> 16x64 output tile; K-loop over D in steps of 32.
// ---------------------------------------------------------------------------
__global__ __launch_bounds__(256) void out_proj(
    const _Float16* __restrict__ a, const _Float16* __restrict__ w,
    const float* __restrict__ bias, float* __restrict__ out)
{
  const int lane = threadIdx.x & 31;
  const int wave = threadIdx.x >> 5;
  const int m    = lane & 15;
  const int hf   = lane >> 4;

  const int wid  = blockIdx.x * 8 + wave;   // 8192 waves total
  const int row0 = (wid >> 4) * 16;         // 512 row tiles
  const int col0 = (wid & 15) * 64;         // 16 col tiles

  v8f acc[4];
#pragma unroll
  for (int c = 0; c < 4; ++c) acc[c] = vzero8();

  for (int kk = 0; kk < D_; kk += 32) {
    const _Float16* ap = a + (size_t)(row0 + m) * D_ + kk + hf * 8;
    v16h af = ldfrag(ap, ap + 16);
#pragma unroll
    for (int c = 0; c < 4; ++c) {
      // B element [kdim][n] = W[col0+c*16+n][kdim]: contiguous along kdim
      const _Float16* wp = w + (size_t)(col0 + c * 16 + m) * D_ + kk + hf * 8;
      v16h bf = ldfrag(wp, wp + 16);
      acc[c] = wmma16x16x32(af, bf, acc[c]);
    }
  }

#pragma unroll
  for (int c = 0; c < 4; ++c) {
    float bj = bias[col0 + c * 16 + m];
#pragma unroll
    for (int r = 0; r < 8; ++r) {
      int row = row0 + 8 * hf + r;
      int col = col0 + c * 16 + m;
      out[(size_t)row * D_ + col] = acc[c][r] + bj;
    }
  }
}

// ---------------------------------------------------------------------------
extern "C" void kernel_launch(void* const* d_in, const int* in_sizes, int n_in,
                              void* d_out, int out_size, void* d_ws, size_t ws_size,
                              hipStream_t stream)
{
  const float* x  = (const float*)d_in[0];
  const float* tq = (const float*)d_in[1];
  const float* tk = (const float*)d_in[2];
  const float* tv = (const float*)d_in[3];
  const float* W  = (const float*)d_in[4];
  const float* bb = (const float*)d_in[5];
  float* out = (float*)d_out;

  const size_t QKV_ELEMS = (size_t)BH_ * N_ * HD_;       // 8,388,608
  const size_t QKV_BYTES = QKV_ELEMS * sizeof(_Float16); // 16 MB

  char* ws = (char*)d_ws;
  _Float16* qf  = (_Float16*)(ws + 0 * QKV_BYTES);
  _Float16* kf  = (_Float16*)(ws + 1 * QKV_BYTES);
  _Float16* vtf = (_Float16*)(ws + 2 * QKV_BYTES);
  _Float16* of  = (_Float16*)(ws + 3 * QKV_BYTES);
  _Float16* wf  = (_Float16*)(ws + 4 * QKV_BYTES);       // D*D f16 = 2 MB

  // 1) cos projections -> f16 (Q pre-scaled, V transposed) + W -> f16
  qkv_pack<<<(B_ * N_ * D_) / 256, 256, 0, stream>>>(x, tq, tk, tv, qf, kf, vtf);
  w_pack<<<(D_ * D_) / 256, 256, 0, stream>>>(W, wf);

  // 2) attention: BH * (N/128) blocks, 8 waves each
  qattn_flash<<<BH_ * (N_ / 128), 256, 0, stream>>>(qf, kf, vtf, of);

  // 3) output projection: (BN/16)*(D/64) = 8192 waves -> 1024 blocks
  out_proj<<<1024, 256, 0, stream>>>(of, wf, bb, out);
}
